// ClassTransformerLayer_26903675142339
// MI455X (gfx1250) — matrix operand order repaired and checked by reading, not compiled, not run on hardware
//
#include <hip/hip_runtime.h>
#include <hip/hip_bf16.h>
#include <math.h>

#define HID 128
#define GD  128
#define NH  8
#define DH  16
#define PAD 128
#define TT  77
#define LTOK 2
#define HW  24
#define SP  (HW*HW)
#define NB  4

typedef __attribute__((ext_vector_type(16))) _Float16 v16h;
typedef __attribute__((ext_vector_type(8)))  _Float16 v8h;
typedef __attribute__((ext_vector_type(8)))  float    v8f;
typedef __attribute__((ext_vector_type(4)))  unsigned int u32x4;
typedef __attribute__((ext_vector_type(8)))  int i32x8;
typedef __attribute__((ext_vector_type(4)))  int i32x4;

__device__ __forceinline__ float elu1(float x) {
    return x > 0.f ? x + 1.f : __expf(x);
}

// A fragment (16x32 f16): lane<16 -> halves 0..7 = K[k0..k0+7], 8..15 = K[k0+16..23]
//                         lane>=16 -> halves 0..7 = K[k0+8..15], 8..15 = K[k0+24..31]
__device__ __forceinline__ v16h load_a_frag(const _Float16* base, int row, int rs,
                                            int k0, int lane) {
    int off = (lane < 16) ? 0 : 8;
    v8h lo = *(const v8h*)(base + row * rs + k0 + off);
    v8h hi = *(const v8h*)(base + row * rs + k0 + 16 + off);
    return __builtin_shufflevector(lo, hi, 0,1,2,3,4,5,6,7,8,9,10,11,12,13,14,15);
}

// B fragment (32x16 f16): lane L holds column n = n0+L%16; lane<16 -> K[k0..7],K[k0+16..23], lane>=16 -> K[k0+8..15],K[k0+24..31]
__device__ __forceinline__ v16h load_b_frag(const _Float16* base, int k0, int rs,
                                            int n0, int lane) {
    int off = (lane < 16) ? 0 : 8;
    const _Float16* p = base + n0 + (lane & 15);
    v16h r;
    for (int s = 0; s < 8; ++s) {
        r[s]     = p[(k0 + off + s) * rs];
        r[s + 8] = p[(k0 + 16 + off + s) * rs];
    }
    return r;
}

// TDM descriptor: copy a 128x128 2-byte-element tile (contiguous, stride 128)
// from global memory into LDS. Issued by one wave; EXEC ignored by TDM.
__device__ __forceinline__ void tdm_load_tile_128x128_h(void* lds_ptr, const void* gptr) {
    unsigned lds = (unsigned)(unsigned long long)(uintptr_t)lds_ptr; // LDS offset in addr[31:0]
    unsigned long long ga = (unsigned long long)(uintptr_t)gptr;
    u32x4 g0;
    g0[0] = 1u;                                        // count=1, user descriptor
    g0[1] = lds;                                       // lds_addr
    g0[2] = (unsigned)(ga & 0xffffffffu);              // global_addr[31:0]
    g0[3] = (unsigned)((ga >> 32) & 0x1ffffffu)        // global_addr[56:32]
          | (2u << 30);                                // type = 2 ("image")
    i32x8 g1;
    g1[0] = (int)(1u << 16);                           // data_size = 1 -> 2 bytes
    g1[1] = (int)(128u << 16);                         // tensor_dim0 = 128 (lo16 in [31:16])
    g1[2] = (int)(128u << 16);                         // tensor_dim0 hi=0, tensor_dim1 lo = 128
    g1[3] = (int)(128u << 16);                         // tensor_dim1 hi=0, tile_dim0 = 128
    g1[4] = 128;                                       // tile_dim1 = 128, tile_dim2 = 0
    g1[5] = 128;                                       // tensor_dim0_stride = 128 (lo32)
    g1[6] = 0;                                         // stride hi, tensor_dim1_stride lo
    g1[7] = 0;
    i32x4 z4 = {0, 0, 0, 0};
    i32x8 z8 = {0, 0, 0, 0, 0, 0, 0, 0};
    __builtin_amdgcn_tensor_load_to_lds(g0, g1, z4, z4, z8, 0);
}

// ---------------------------------------------------------------------------
// Kernel 1: per-batch guidance part of K projection:
//   kg[b][s][c] = sum_i gfull[b][s][i] * wk[128+i][c] + bk[c]
// ---------------------------------------------------------------------------
__global__ __launch_bounds__(256) void kg_precompute(
    const float* __restrict__ guidance, const float* __restrict__ pad_g,
    const float* __restrict__ wk, const float* __restrict__ bk,
    float* __restrict__ kg)
{
    int b = blockIdx.x, tid = threadIdx.x;
    for (int e = tid; e < PAD * HID; e += 256) {
        int s = e >> 7, c = e & 127;
        float acc = bk[c];
        const float* grow = (s < TT) ? (guidance + (b * TT + s) * GD) : pad_g;
        for (int i = 0; i < GD; ++i)
            acc += grow[i] * wk[(HID + i) * HID + c];
        kg[b * PAD * HID + e] = acc;
    }
}

// ---------------------------------------------------------------------------
// Kernel 1b: convert wk_top (rows 0..127) and wv to f16 row-major in workspace
//   whalf[0..16383]      = wk[0:128][:]
//   whalf[16384..32767]  = wv[:][:]
// ---------------------------------------------------------------------------
__global__ __launch_bounds__(256) void w_convert(
    const float* __restrict__ wk, const float* __restrict__ wv,
    _Float16* __restrict__ whalf)
{
    int e = blockIdx.x * 256 + threadIdx.x;
    if (e < HID * HID)
        whalf[e] = (_Float16)wk[e];
    else if (e < 2 * HID * HID)
        whalf[e] = (_Float16)wv[e - HID * HID];
}

// ---------------------------------------------------------------------------
// Kernel 2: one workgroup (256 threads = 8 waves) per spatial position.
// ---------------------------------------------------------------------------
__global__ __launch_bounds__(256) void ctl_main(
    const float* __restrict__ x, const float* __restrict__ query,
    const float* __restrict__ ltok, const float* __restrict__ padtok,
    const float* __restrict__ wq, const float* __restrict__ bq,
    const float* __restrict__ bv,
    const float* __restrict__ w1, const float* __restrict__ b1,
    const float* __restrict__ w2, const float* __restrict__ b2,
    const float* __restrict__ g1, const float* __restrict__ be1,
    const float* __restrict__ g2, const float* __restrict__ be2,
    const float* __restrict__ g3, const float* __restrict__ be3,
    const float* __restrict__ kg, const _Float16* __restrict__ whalf,
    float* __restrict__ out)
{
    __shared__ __align__(16) _Float16 s_xn[PAD * HID];   // LN'd x tokens (row-major)
    __shared__ __align__(16) _Float16 s_wB[HID * HID];   // TDM-staged weight (k x n)
    __shared__ __align__(16) _Float16 s_Kt[HID * PAD];   // elu(K)+1, TRANSPOSED [chan][tok]
    __shared__ __align__(16) _Float16 s_V [PAD * HID];   // V/S, row-major [tok][chan]
    __shared__ float s_qn[LTOK * 256];
    __shared__ float s_Q [LTOK * HID];
    __shared__ float s_Ksum[HID];
    __shared__ float s_KV[NH * DH * DH];
    __shared__ float s_qv[LTOK * HID];
    __shared__ float s_out[HID];
    __shared__ float s_hn[HID];
    __shared__ float s_m1[4 * HID];
    __shared__ float s_stats[2];

    const int tid  = threadIdx.x;
    const int wave = tid >> 5;
    const int lane = tid & 31;
    const int mrow = lane & 15;
    const int half16 = (lane < 16) ? 0 : 1;

    const int n   = blockIdx.x;
    const int b   = n / SP;
    const int pix = n % SP;

    // ---- Phase 1: TDM DMA of wk_top f16 into s_wB, overlapped with:
    //      LN of x tokens (tid 0..127) and LN of q-input (tid 128..129)
    if (wave == 0)
        tdm_load_tile_128x128_h(s_wB, whalf);           // async: TENSORcnt++

    if (tid < PAD) {
        int s = tid;
        float sum = 0.f, sq = 0.f;
        for (int c = 0; c < HID; ++c) {
            float v = (s < TT) ? x[((b * HID + c) * TT + s) * SP + pix] : padtok[c];
            sum += v; sq += v * v;
            s_xn[s * HID + c] = (_Float16)v;            // park raw value (f16)
        }
        float mean = sum * (1.f / HID);
        float var  = sq * (1.f / HID) - mean * mean;
        float rstd = rsqrtf(var + 1e-5f);
        for (int c = 0; c < HID; ++c) {
            float v = (float)s_xn[s * HID + c];
            s_xn[s * HID + c] = (_Float16)((v - mean) * rstd * g2[c] + be2[c]);
        }
    } else if (tid < 128 + LTOK) {
        int l = tid - 128;
        float sum = 0.f, sq = 0.f;
        for (int i = 0; i < HID + GD; ++i) {
            float v = (i < HID) ? ltok[l * HID + i]
                                : query[(b * GD + (i - HID)) * SP + pix];
            sum += v; sq += v * v;
        }
        float mean = sum * (1.f / (HID + GD));
        float var  = sq * (1.f / (HID + GD)) - mean * mean;
        float rstd = rsqrtf(var + 1e-5f);
        for (int i = 0; i < HID + GD; ++i) {
            float v = (i < HID) ? ltok[l * HID + i]
                                : query[(b * GD + (i - HID)) * SP + pix];
            s_qn[l * 256 + i] = (v - mean) * rstd * g1[i] + be1[i];
        }
    }
    if (wave == 0)
        __builtin_amdgcn_s_wait_tensorcnt((short)0);    // wk tile landed
    __syncthreads();

    // ---- Phase 2: K = xn @ wk_top + kg[b]; elu+1; store transposed into s_Kt
    for (int t = wave; t < 64; t += 8) {
        int tm = t >> 3, tn = t & 7;
        v8f c;
        for (int r = 0; r < 8; ++r)
            c[r] = kg[b * PAD * HID + (tm * 16 + r + half16 * 8) * HID + tn * 16 + mrow];
        for (int kk = 0; kk < 4; ++kk) {
            v16h a  = load_a_frag(s_xn, tm * 16 + mrow, HID, kk * 32, lane);
            v16h bb = load_b_frag(s_wB, kk * 32, HID, tn * 16, lane);
            c = __builtin_amdgcn_wmma_f32_16x16x32_f16(false, a, false, bb,
                                                       (short)0, c, false, false);
        }
        for (int r = 0; r < 8; ++r) {
            int m  = tm * 16 + r + half16 * 8;
            int nn = tn * 16 + mrow;
            s_Kt[nn * PAD + m] = (_Float16)elu1(c[r]);
        }
    }
    __syncthreads();

    // ---- Phase 3: TDM DMA of wv f16 into s_wB, overlapped with
    //      Q = elu(qn @ wq + bq) + 1   (2 x 128 outputs, 1 per thread)
    if (wave == 0)
        tdm_load_tile_128x128_h(s_wB, whalf + HID * HID);
    {
        int l = tid >> 7, co = tid & 127;
        float acc = bq[co];
        for (int i = 0; i < HID + GD; ++i)
            acc += s_qn[l * 256 + i] * wq[i * HID + co];
        s_Q[l * HID + co] = elu1(acc);
    }
    if (wave == 0)
        __builtin_amdgcn_s_wait_tensorcnt((short)0);    // wv tile landed
    __syncthreads();

    // ---- Phase 4: V = (xn @ wv + bv) / PAD, row-major into s_V
    for (int t = wave; t < 64; t += 8) {
        int tm = t >> 3, tn = t & 7;
        float bvn = bv[tn * 16 + mrow];
        v8f c = {bvn, bvn, bvn, bvn, bvn, bvn, bvn, bvn};
        for (int kk = 0; kk < 4; ++kk) {
            v16h a  = load_a_frag(s_xn, tm * 16 + mrow, HID, kk * 32, lane);
            v16h bb = load_b_frag(s_wB, kk * 32, HID, tn * 16, lane);
            c = __builtin_amdgcn_wmma_f32_16x16x32_f16(false, a, false, bb,
                                                       (short)0, c, false, false);
        }
        for (int r = 0; r < 8; ++r) {
            int m  = tm * 16 + r + half16 * 8;
            int nn = tn * 16 + mrow;
            s_V[m * HID + nn] = (_Float16)(c[r] * (1.f / PAD));
        }
    }
    __syncthreads();

    // ---- Phase 5: Ksum over tokens (rows of s_Kt are contiguous)
    if (tid < HID) {
        float s = 0.f;
        for (int j = 0; j < PAD; ++j) s += (float)s_Kt[tid * PAD + j];
        s_Ksum[tid] = s;
    }
    __syncthreads();

    // ---- Phase 6: KV[h] = K_h^T @ V_h  (one head per wave, 4 WMMA k-steps)
    {
        int hh = wave;
        v8f c = {0.f, 0.f, 0.f, 0.f, 0.f, 0.f, 0.f, 0.f};
        for (int kk = 0; kk < 4; ++kk) {
            v16h a  = load_a_frag(s_Kt, hh * 16 + mrow, PAD, kk * 32, lane);
            v16h bb = load_b_frag(s_V, kk * 32, HID, hh * 16, lane);
            c = __builtin_amdgcn_wmma_f32_16x16x32_f16(false, a, false, bb,
                                                       (short)0, c, false, false);
        }
        for (int r = 0; r < 8; ++r)
            s_KV[hh * (DH * DH) + (r + half16 * 8) * DH + mrow] = c[r];
    }
    __syncthreads();

    // ---- Phase 7: qv[l,h,n] = (Q . KV) * Z * PAD ; then mean over l
    {
        int l = tid >> 7, rem = tid & 127;
        int hh = rem >> 4, nn2 = rem & 15;
        float z = 0.f, num = 0.f;
        for (int d = 0; d < DH; ++d) {
            float qd = s_Q[l * HID + hh * 16 + d];
            z   += qd * s_Ksum[hh * 16 + d];
            num += qd * s_KV[hh * (DH * DH) + d * DH + nn2];
        }
        s_qv[l * HID + rem] = num * (1.f / (z + 1e-6f)) * (float)PAD;
    }
    __syncthreads();
    if (tid < HID) s_out[tid] = 0.5f * (s_qv[tid] + s_qv[HID + tid]);
    __syncthreads();

    // ---- Phase 8: LayerNorm of out (stats by thread 0; tiny)
    if (tid == 0) {
        float sum = 0.f, sq = 0.f;
        for (int c = 0; c < HID; ++c) { float v = s_out[c]; sum += v; sq += v * v; }
        float mean = sum * (1.f / HID);
        float var  = sq * (1.f / HID) - mean * mean;
        s_stats[0] = mean;
        s_stats[1] = rsqrtf(var + 1e-5f);
    }
    __syncthreads();
    if (tid < HID)
        s_hn[tid] = (s_out[tid] - s_stats[0]) * s_stats[1] * g3[tid] + be3[tid];
    __syncthreads();

    // ---- Phase 9: hidden = relu(hn @ w1 + b1)
    for (int j = tid; j < 4 * HID; j += 256) {
        float acc = b1[j];
        for (int i = 0; i < HID; ++i) acc += s_hn[i] * w1[i * (4 * HID) + j];
        s_m1[j] = fmaxf(acc, 0.f);
    }
    __syncthreads();

    // ---- Phase 10: out += hidden @ w2 + b2 ; write (B,HID,H,W)
    if (tid < HID) {
        float acc = b2[tid];
        for (int j = 0; j < 4 * HID; ++j) acc += s_m1[j] * w2[j * HID + tid];
        out[(b * HID + tid) * SP + pix] = s_out[tid] + acc;
    }
}

extern "C" void kernel_launch(void* const* d_in, const int* in_sizes, int n_in,
                              void* d_out, int out_size, void* d_ws, size_t ws_size,
                              hipStream_t stream) {
    const float* x        = (const float*)d_in[0];
    const float* guidance = (const float*)d_in[1];
    const float* query    = (const float*)d_in[2];
    const float* ltok     = (const float*)d_in[3];
    const float* padtok   = (const float*)d_in[4];
    const float* padg     = (const float*)d_in[5];
    const float* wq = (const float*)d_in[6];  const float* bq = (const float*)d_in[7];
    const float* wk = (const float*)d_in[8];  const float* bk = (const float*)d_in[9];
    const float* wv = (const float*)d_in[10]; const float* bv = (const float*)d_in[11];
    const float* w1 = (const float*)d_in[12]; const float* b1 = (const float*)d_in[13];
    const float* w2 = (const float*)d_in[14]; const float* b2 = (const float*)d_in[15];
    const float* g1 = (const float*)d_in[16]; const float* be1 = (const float*)d_in[17];
    const float* g2 = (const float*)d_in[18]; const float* be2 = (const float*)d_in[19];
    const float* g3 = (const float*)d_in[20]; const float* be3 = (const float*)d_in[21];
    float* outp = (float*)d_out;

    float*     kgws  = (float*)d_ws;                         // 4*128*128 f32 = 256 KB
    _Float16*  whalf = (_Float16*)((char*)d_ws + NB * PAD * HID * sizeof(float));
                                                             // 2*128*128 f16 = 64 KB

    kg_precompute<<<NB, 256, 0, stream>>>(guidance, padg, wk, bk, kgws);
    w_convert<<<(2 * HID * HID + 255) / 256, 256, 0, stream>>>(wk, wv, whalf);
    ctl_main<<<NB * SP, 256, 0, stream>>>(x, query, ltok, padtok, wq, bq, bv,
                                          w1, b1, w2, b2, g1, be1, g2, be2, g3, be3,
                                          kgws, whalf, outp);
}
